// _MRGNNLayer_90941637526228
// MI455X (gfx1250) — matrix operand order repaired and checked by reading, not compiled, not run on hardware
//
#include <hip/hip_runtime.h>
#include <cstdint>
#include <cstddef>

// ---------------------------------------------------------------------------
// Problem constants (from reference): B=4, N=2048, D=128, E=65536, EF=16
// ---------------------------------------------------------------------------
constexpr int Bc  = 4;
constexpr int Nc  = 2048;
constexpr int Dc  = 128;
constexpr int Ec  = 65536;
constexpr int EFc = 16;

constexpr int H1   = 256;               // hidden width; also per-wave scratch stride
constexpr int NKC1 = 5,  NNT1 = 16;     // GEMM1: K=160 (5 chunks of 32), N=256 (16 tiles)
constexpr int NKC2 = 8,  NNT2 = 8;      // GEMM2: K=256, N=128
constexpr int W1F_ELEMS = NNT1 * NKC1 * 32 * 16;   // 40960 bf16
constexpr int W2F_ELEMS = NNT2 * NKC2 * 32 * 16;   // 32768 bf16

constexpr int WAVES = 16;               // waves per block (512 threads)
constexpr int THREADS = WAVES * 32;

typedef __attribute__((ext_vector_type(16))) __bf16 v16bf;
typedef __attribute__((ext_vector_type(8)))  float  v8f;

union Frag { v16bf v; uint4 u[2]; };

// ---------------------------------------------------------------------------
// Utility kernels
// ---------------------------------------------------------------------------
__global__ void zero_f32(float* __restrict__ p, int n) {
  int i = blockIdx.x * blockDim.x + threadIdx.x;
  if (i < n) p[i] = 0.f;
}

__global__ void cvt_bf16(const float* __restrict__ s, __bf16* __restrict__ d, int n) {
  int i = blockIdx.x * blockDim.x + threadIdx.x;
  if (i < n) d[i] = (__bf16)s[i];
}

// Pack f32 weight [K x Ncols] (row-major) into WMMA B-fragment-ready bf16 layout:
// linear index = ((nt*nKc + kc)*32 + lane)*16 + e
// holds B[K = kc*32 + (e<8 ? e + 8*(lane>=16) : 16 + (e-8) + 8*(lane>=16))][N = nt*16 + (lane&15)], zero-padded past Kreal.
__global__ void pack_wfrag(const float* __restrict__ w, __bf16* __restrict__ dst,
                           int Kreal, int Ncols, int nKc, int total) {
  int idx = blockIdx.x * blockDim.x + threadIdx.x;
  if (idx >= total) return;
  int e    = idx & 15;
  int lane = (idx >> 4) & 31;
  int rest = idx >> 9;
  int kc   = rest % nKc;
  int nt   = rest / nKc;
  int k = kc * 32 + ((e < 8) ? e : 8 + e) + ((lane >> 4) << 3);
  int n = nt * 16 + (lane & 15);
  float v = (k < Kreal) ? w[k * Ncols + n] : 0.f;
  dst[idx] = (__bf16)v;
}

__global__ void count_edges(const int* __restrict__ dst, float* __restrict__ cnt, int n) {
  int i = blockIdx.x * blockDim.x + threadIdx.x;
  if (i < n) atomicAdd(cnt + dst[i], 1.f);
}

// ---------------------------------------------------------------------------
// Fused edge MLP: per wave a 16-edge tile.
//   X(16x160)=[h_src|ea|0] ; H = relu(X@W1+b1) (16x256) ; M = H@W2+b2 (16x128)
//   atomic scatter-add of M rows into agg[b, dst, :]
// X staging and hidden staging share one per-wave 16x256 bf16 region: X (cols
// 0..159) is fully loaded into A1 register fragments before the first hidden
// store, and same-wave LDS ops are in-order (DScnt), so the overwrite is safe.
// Grid: (E/256, B), block 512 (16 waves). Dynamic LDS = 272 KB -> 4 waves/SIMD.
// ---------------------------------------------------------------------------
__global__ __launch_bounds__(THREADS)
void edge_mlp(const __bf16* __restrict__ hbf, const int* __restrict__ ei,
              const __bf16* __restrict__ eabf, const __bf16* __restrict__ wf,
              const float* __restrict__ b1, const float* __restrict__ b2,
              float* __restrict__ agg) {
  extern __shared__ __attribute__((aligned(16))) char smem[];
  __bf16* sw1 = (__bf16*)smem;                                            // 80 KB
  __bf16* sw2 = (__bf16*)(smem + (size_t)W1F_ELEMS * 2);                  // 64 KB
  __bf16* stb = (__bf16*)(smem + (size_t)(W1F_ELEMS + W2F_ELEMS) * 2);    // 16 x 8 KB

  const int tid = threadIdx.x;

  // Cooperative staging of both packed weight blocks (contiguous in global).
  {
    const uint4* src = (const uint4*)wf;
    uint4* dstv = (uint4*)smem;
    constexpr int NV = (W1F_ELEMS + W2F_ELEMS) / 8;   // 9216 uint4
    for (int i = tid; i < NV; i += THREADS) dstv[i] = src[i];
  }
  __syncthreads();

  const int wave = tid >> 5, lane = tid & 31;
  const int row = lane & 15, hi = lane >> 4;   // A-layout: M=lane&15, K-half select
  const int tileBase = (blockIdx.x * WAVES + wave) * 16;
  const int b = blockIdx.y;
  __bf16* st = stb + (size_t)wave * 16 * H1;   // per-wave 16x256 scratch (X then H)

  // ---- gather X tile: [h[b, src(e), 0:128] | ea[e, 0:16] | zeros(16)] ----
  {
    int e = tileBase + row;
    int s = ei[e];                                        // src node
    const uint4* hrow = (const uint4*)(hbf + ((size_t)b * Nc + s) * Dc);
    uint4* xrow = (uint4*)(st + (size_t)row * H1);
#pragma unroll
    for (int i = 0; i < 8; ++i) xrow[hi * 8 + i] = hrow[hi * 8 + i];
    if (hi == 0) {
      const uint4* er = (const uint4*)(eabf + (size_t)e * EFc);
      xrow[16] = er[0]; xrow[17] = er[1];
    } else {
      uint4 z; z.x = z.y = z.z = z.w = 0u;
      xrow[18] = z; xrow[19] = z;
    }
    if (e + 4096 < Ec) __builtin_prefetch(eabf + (size_t)(e + 4096) * EFc, 0, 1);
  }

  const int koff = hi * 8, nlo = lane & 15;

  // ---- A fragments for GEMM1 (ISA 16-bit A 16x32 layout, two b128 LDS reads) ----
  Frag a1[NKC1];
#pragma unroll
  for (int kc = 0; kc < NKC1; ++kc) {
    a1[kc].u[0] = *(const uint4*)(st + (size_t)row * H1 + kc * 32 + koff);
    a1[kc].u[1] = *(const uint4*)(st + (size_t)row * H1 + kc * 32 + 16 + koff);
  }

  // ---- GEMM1: 16 N-tiles x 5 K-chunks of v_wmma_f32_16x16x32_bf16 ----
#pragma unroll 1
  for (int nt = 0; nt < NNT1; ++nt) {
    v8f acc = {};
#pragma unroll
    for (int kc = 0; kc < NKC1; ++kc) {
      Frag bw;
      const uint4* p = (const uint4*)(sw1 + (size_t)((nt * NKC1 + kc) * 32 + lane) * 16);
      bw.u[0] = p[0]; bw.u[1] = p[1];
      acc = __builtin_amdgcn_wmma_f32_16x16x32_bf16(false, a1[kc].v, false, bw.v,
                                                    (short)0, acc, false, false);
    }
    float bias = b1[nt * 16 + nlo];
#pragma unroll
    for (int j = 0; j < 8; ++j) {                 // D layout: lane<16 M=j, lane>=16 M=j+8
      float v = acc[j] + bias;
      v = v > 0.f ? v : 0.f;                      // relu
      st[(size_t)(j + hi * 8) * H1 + nt * 16 + nlo] = (__bf16)v;
    }
  }

  // ---- A fragments for GEMM2 from hidden tile ----
  Frag a2[NKC2];
#pragma unroll
  for (int kc = 0; kc < NKC2; ++kc) {
    a2[kc].u[0] = *(const uint4*)(st + (size_t)row * H1 + kc * 32 + koff);
    a2[kc].u[1] = *(const uint4*)(st + (size_t)row * H1 + kc * 32 + 16 + koff);
  }

  // destination node id per output row (rows 0..15 held by lanes 0..15)
  int dv = ei[Ec + tileBase + row];
  int dmj[8];
#pragma unroll
  for (int j = 0; j < 8; ++j) dmj[j] = __shfl(dv, j + hi * 8, 32);

  // ---- GEMM2 + bias + atomic segment-sum scatter ----
#pragma unroll 1
  for (int nt = 0; nt < NNT2; ++nt) {
    v8f acc = {};
#pragma unroll
    for (int kc = 0; kc < NKC2; ++kc) {
      Frag bw;
      const uint4* p = (const uint4*)(sw2 + (size_t)((nt * NKC2 + kc) * 32 + lane) * 16);
      bw.u[0] = p[0]; bw.u[1] = p[1];
      acc = __builtin_amdgcn_wmma_f32_16x16x32_bf16(false, a2[kc].v, false, bw.v,
                                                    (short)0, acc, false, false);
    }
    float bias = b2[nt * 16 + nlo];
#pragma unroll
    for (int j = 0; j < 8; ++j) {
      float v = acc[j] + bias;
      atomicAdd(agg + ((size_t)b * Nc + dmj[j]) * Dc + nt * 16 + nlo, v);
    }
  }
}

// ---------------------------------------------------------------------------
// Final node update: upd = [agg_t/cnt_t | agg_d/cnt_d] @ wu + bu ; y=h+upd ; LN
// One wave per node; lane owns 4 output channels.
// ---------------------------------------------------------------------------
__global__ __launch_bounds__(256)
void final_update(const float* __restrict__ h, const float* __restrict__ agg_t,
                  const float* __restrict__ agg_d, const float* __restrict__ cnt_t,
                  const float* __restrict__ cnt_d, const float* __restrict__ wu,
                  const float* __restrict__ bu, const float* __restrict__ gamma,
                  const float* __restrict__ beta, float* __restrict__ out) {
  __shared__ float sbuf[8][H1];
  const int wave = threadIdx.x >> 5, lane = threadIdx.x & 31;
  const int node = blockIdx.x * 8 + wave;       // B*N nodes total
  const int b = node >> 11, n = node & (Nc - 1);
  const float it = 1.f / fmaxf(cnt_t[n], 1.f);
  const float idv = 1.f / fmaxf(cnt_d[n], 1.f);
  const size_t base = ((size_t)b * Nc + n) * Dc;

#pragma unroll
  for (int i = 0; i < 8; ++i) {
    int k = lane * 8 + i;
    sbuf[wave][k] = (k < Dc) ? agg_t[base + k] * it : agg_d[base + k - Dc] * idv;
  }

  float upd[4];
#pragma unroll
  for (int j = 0; j < 4; ++j) upd[j] = bu[lane + 32 * j];
  for (int k = 0; k < H1; ++k) {
    float xv = sbuf[wave][k];
#pragma unroll
    for (int j = 0; j < 4; ++j) upd[j] = fmaf(xv, wu[k * Dc + lane + 32 * j], upd[j]);
  }

  float y[4], s = 0.f;
#pragma unroll
  for (int j = 0; j < 4; ++j) { y[j] = h[base + lane + 32 * j] + upd[j]; s += y[j]; }
#pragma unroll
  for (int off = 16; off > 0; off >>= 1) s += __shfl_xor(s, off, 32);
  float mu = s * (1.f / Dc);
  float vs = 0.f;
#pragma unroll
  for (int j = 0; j < 4; ++j) { float d0 = y[j] - mu; vs += d0 * d0; }
#pragma unroll
  for (int off = 16; off > 0; off >>= 1) vs += __shfl_xor(vs, off, 32);
  float inv = rsqrtf(vs * (1.f / Dc) + 1e-5f);
#pragma unroll
  for (int j = 0; j < 4; ++j) {
    int dch = lane + 32 * j;
    out[base + dch] = (y[j] - mu) * inv * gamma[dch] + beta[dch];
  }
}

// ---------------------------------------------------------------------------
// Host orchestration
// ---------------------------------------------------------------------------
extern "C" void kernel_launch(void* const* d_in, const int* in_sizes, int n_in,
                              void* d_out, int out_size, void* d_ws, size_t ws_size,
                              hipStream_t stream) {
  (void)in_sizes; (void)n_in; (void)out_size; (void)ws_size;
  const float* h   = (const float*)d_in[0];
  const int*   tei = (const int*)d_in[1];
  const float* tea = (const float*)d_in[2];
  const int*   dei = (const int*)d_in[3];
  const float* dea = (const float*)d_in[4];
  const float* wt1 = (const float*)d_in[5];
  const float* bt1 = (const float*)d_in[6];
  const float* wt2 = (const float*)d_in[7];
  const float* bt2 = (const float*)d_in[8];
  const float* wd1 = (const float*)d_in[9];
  const float* bd1 = (const float*)d_in[10];
  const float* wd2 = (const float*)d_in[11];
  const float* bd2 = (const float*)d_in[12];
  const float* wu  = (const float*)d_in[13];
  const float* bu  = (const float*)d_in[14];
  const float* gam = (const float*)d_in[15];
  const float* bet = (const float*)d_in[16];
  float* out = (float*)d_out;
  char* ws = (char*)d_ws;

  size_t off = 0;
  float* agg_t = (float*)(ws + off); off += (size_t)Bc * Nc * Dc * 4;
  float* agg_d = (float*)(ws + off); off += (size_t)Bc * Nc * Dc * 4;
  float* cnt_t = (float*)(ws + off); off += (size_t)Nc * 4;
  float* cnt_d = (float*)(ws + off); off += (size_t)Nc * 4;
  const int zeroFloats = (int)(off / 4);
  __bf16* hbf  = (__bf16*)(ws + off); off += (size_t)Bc * Nc * Dc * 2;
  __bf16* teab = (__bf16*)(ws + off); off += (size_t)Ec * EFc * 2;
  __bf16* deab = (__bf16*)(ws + off); off += (size_t)Ec * EFc * 2;
  __bf16* wf_t = (__bf16*)(ws + off); off += (size_t)(W1F_ELEMS + W2F_ELEMS) * 2;
  __bf16* wf_d = (__bf16*)(ws + off); off += (size_t)(W1F_ELEMS + W2F_ELEMS) * 2;

  zero_f32<<<(zeroFloats + 255) / 256, 256, 0, stream>>>((float*)ws, zeroFloats);

  const int nh = Bc * Nc * Dc;
  cvt_bf16<<<(nh + 255) / 256, 256, 0, stream>>>(h, hbf, nh);
  const int ne = Ec * EFc;
  cvt_bf16<<<(ne + 255) / 256, 256, 0, stream>>>(tea, teab, ne);
  cvt_bf16<<<(ne + 255) / 256, 256, 0, stream>>>(dea, deab, ne);

  pack_wfrag<<<(W1F_ELEMS + 255) / 256, 256, 0, stream>>>(wt1, wf_t, 144, 256, NKC1, W1F_ELEMS);
  pack_wfrag<<<(W2F_ELEMS + 255) / 256, 256, 0, stream>>>(wt2, wf_t + W1F_ELEMS, 256, 128, NKC2, W2F_ELEMS);
  pack_wfrag<<<(W1F_ELEMS + 255) / 256, 256, 0, stream>>>(wd1, wf_d, 144, 256, NKC1, W1F_ELEMS);
  pack_wfrag<<<(W2F_ELEMS + 255) / 256, 256, 0, stream>>>(wd2, wf_d + W1F_ELEMS, 256, 128, NKC2, W2F_ELEMS);

  count_edges<<<(Ec + 255) / 256, 256, 0, stream>>>(tei + Ec, cnt_t, Ec);
  count_edges<<<(Ec + 255) / 256, 256, 0, stream>>>(dei + Ec, cnt_d, Ec);

  // 80KB (W1 frags) + 64KB (W2 frags) + 16 waves * 8KB scratch = 272 KB
  constexpr unsigned SMEM = (W1F_ELEMS + W2F_ELEMS) * 2 + WAVES * 16 * H1 * 2;
  dim3 eg(Ec / (WAVES * 16), Bc);
  edge_mlp<<<eg, THREADS, SMEM, stream>>>(hbf, tei, teab, wf_t, bt1, bt2, agg_t);
  edge_mlp<<<eg, THREADS, SMEM, stream>>>(hbf, dei, deab, wf_d, bd1, bd2, agg_d);

  final_update<<<(Bc * Nc) / 8, 256, 0, stream>>>(h, agg_t, agg_d, cnt_t, cnt_d,
                                                  wu, bu, gam, bet, out);
}
